// HistoricalPriorRangeQDSModel_46110768890441
// MI455X (gfx1250) — matrix operand order repaired and checked by reading, not compiled, not run on hardware
//
#include <hip/hip_runtime.h>

typedef __attribute__((ext_vector_type(16))) _Float16 v16h;
typedef __attribute__((ext_vector_type(8)))  _Float16 v8h;
typedef __attribute__((ext_vector_type(8)))  float    v8f;

#define NQ     16384      // 8*2048 query points
#define NF     20000      // features / targets
#define DIM    64
#define KNN    32
#define WAVES  8
#define NTILES (NF / 16)  // 1250, exact

// workspace layout (bytes); total ~4.8 MB
#define OFF_QF16 0u
#define OFF_FF16 (NQ * DIM * 2u)            // 2,097,152
#define OFF_QN   (OFF_FF16 + NF * DIM * 2u) // 4,657,152
#define OFF_FN   (OFF_QN + NQ * 4u)         // 4,722,688

// ---------------------------------------------------------------------------
// Prologue: f32 -> f16 conversion + row squared-norms. One wave per 64-dim row.
// Rows [0, NF) = features, rows [NF, NF+NQ) = points. (NF+NQ) % 8 == 0.
// ---------------------------------------------------------------------------
__global__ __launch_bounds__(256) void prep_kernel(
    const float* __restrict__ points, const float* __restrict__ features,
    _Float16* __restrict__ qf16, _Float16* __restrict__ ff16,
    float* __restrict__ qn, float* __restrict__ fn) {
  const int lane = threadIdx.x & 31;
  const int wave = threadIdx.x >> 5;
  const int row  = blockIdx.x * 8 + wave;  // 0 .. NF+NQ-1

  const float* src;
  _Float16* dst;
  float* nrm;
  if (row < NF) {
    src = features + (size_t)row * DIM;
    dst = ff16 + (size_t)row * DIM;
    nrm = fn + row;
  } else {
    const int r = row - NF;
    src = points + (size_t)r * DIM;
    dst = qf16 + (size_t)r * DIM;
    nrm = qn + r;
  }
  const float x0 = src[lane * 2 + 0];
  const float x1 = src[lane * 2 + 1];
  float s = x0 * x0 + x1 * x1;
#pragma unroll
  for (int off = 16; off > 0; off >>= 1) s += __shfl_xor(s, off);
  dst[lane * 2 + 0] = (_Float16)x0;
  dst[lane * 2 + 1] = (_Float16)x1;
  if (lane == 0) *nrm = s;
}

// ---------------------------------------------------------------------------
// Fragment loaders (16-bit operands, wave32); contiguous k order, same in A and B.
// A (16x32, MxK): lane holds row m=lane&15; halves 0-15 <- K = 16*hi + 0..15
//                 (hi = lane>>4).
// B (32x16, KxN): lane holds col n=lane&15; halves 0-15 <- K = 16*hi + 0..15.
// ---------------------------------------------------------------------------
__device__ __forceinline__ v16h load_a_frag(const _Float16* __restrict__ row,
                                            int kc, int hi) {
  return *(const v16h*)(row + kc + 16 * hi);
}

__device__ __forceinline__ v16h load_b_frag(const _Float16* __restrict__ row,
                                            int kc, int hi) {
  return *(const v16h*)(row + kc + 16 * hi);
}

// ---------------------------------------------------------------------------
// Distributed top-32 insertion: 32 slots = 32 lanes, one (d2, target) per lane.
// tau is the current worst (max) of the 32 slots, kept >= true max (safe filter).
// ---------------------------------------------------------------------------
__device__ __forceinline__ void wave_insert(float& bd, float& bt, float& tau,
                                            float d2, float tv, bool half_ok,
                                            int lane) {
  const bool ok = half_ok && (d2 < tau);
  unsigned long long mask = __ballot(ok);
  if (mask) {
    do {
      const int src = __ffsll((long long)mask) - 1;
      mask &= mask - 1;
      const float cv = __shfl(d2, src);
      const float ct = __shfl(tv, src);
      // arg-max of slot values across the wave (tie -> lowest lane)
      float md = bd;
      int ml = lane;
#pragma unroll
      for (int off = 16; off > 0; off >>= 1) {
        const float od = __shfl_xor(md, off);
        const int   ol = __shfl_xor(ml, off);
        if (od > md || (od == md && ol < ml)) { md = od; ml = ol; }
      }
      if (cv < md && lane == ml) { bd = cv; bt = ct; }
    } while (mask);
    // refresh tau (exact max of current slots)
    float m = bd;
#pragma unroll
    for (int off = 16; off > 0; off >>= 1) m = fmaxf(m, __shfl_xor(m, off));
    tau = m;
  }
}

// ---------------------------------------------------------------------------
// Main fused kernel: 16 queries per workgroup, 8 waves stride the 1250
// feature tiles; WMMA distances + in-register top-32; LDS merge at the end.
// ---------------------------------------------------------------------------
__global__ __launch_bounds__(256) void knn_wmma_kernel(
    const _Float16* __restrict__ qf, const _Float16* __restrict__ ff,
    const float* __restrict__ qn, const float* __restrict__ fn,
    const float* __restrict__ targets, float* __restrict__ out) {
  const int lane  = threadIdx.x & 31;
  const int wave  = threadIdx.x >> 5;
  const int hi    = lane >> 4;
  const int lm    = lane & 15;
  const int qbase = blockIdx.x * 16;

  __shared__ float s_qn[16];
  __shared__ float s_d[16][WAVES][KNN];
  __shared__ float s_t[16][WAVES][KNN];

  if (threadIdx.x < 16) s_qn[threadIdx.x] = qn[qbase + threadIdx.x];
  __syncthreads();

  // A tile (queries) loaded once; two K-chunks of 32
  const _Float16* qrow = qf + (size_t)(qbase + lm) * DIM;
  const v16h a0 = load_a_frag(qrow, 0, hi);
  const v16h a1 = load_a_frag(qrow, 32, hi);

  float bd[16], bt[16], tau[16];
#pragma unroll
  for (int q = 0; q < 16; ++q) { bd[q] = 3.0e38f; bt[q] = 0.0f; tau[q] = 3.0e38f; }

  for (int tile = wave; tile < NTILES; tile += WAVES) {
    const int fidx = tile * 16 + lm;
    const _Float16* frow = ff + (size_t)fidx * DIM;
    const v16h b0 = load_b_frag(frow, 0, hi);
    const v16h b1 = load_b_frag(frow, 32, hi);
    const float fnv = fn[fidx];
    const float tv  = targets[fidx];

    v8f acc = {};
    acc = __builtin_amdgcn_wmma_f32_16x16x32_f16(false, a0, false, b0,
                                                 (short)0, acc, false, false);
    acc = __builtin_amdgcn_wmma_f32_16x16x32_f16(false, a1, false, b1,
                                                 (short)0, acc, false, false);

    // D layout: VGPR r, lanes 0-15 -> query m=r, lanes 16-31 -> m=r+8; n=lane&15
#pragma unroll
    for (int r = 0; r < 8; ++r) {
      const float d2 = fmaxf(s_qn[r + 8 * hi] + fnv - 2.0f * acc[r], 0.0f);
      wave_insert(bd[r],     bt[r],     tau[r],     d2, tv, hi == 0, lane);
      wave_insert(bd[r + 8], bt[r + 8], tau[r + 8], d2, tv, hi == 1, lane);
    }
  }

  // dump per-wave lists to LDS
#pragma unroll
  for (int q = 0; q < 16; ++q) {
    s_d[q][wave][lane] = bd[q];
    s_t[q][wave][lane] = bt[q];
  }
  __syncthreads();

  // merge: each wave finishes 2 queries (256 candidates -> smallest 32 -> sum)
#pragma unroll
  for (int j = 0; j < 2; ++j) {
    const int q = wave * 2 + j;
    const float* pd = &s_d[q][0][0];
    const float* pt = &s_t[q][0][0];
    float cd[8], cw[8];
#pragma unroll
    for (int i = 0; i < 8; ++i) {
      cd[i] = pd[lane + 32 * i];
      cw[i] = pt[lane + 32 * i];
    }
    float num = 0.0f, den = 0.0f;
    for (int it = 0; it < KNN; ++it) {
      float mv = cd[0], mt = cw[0];
      int ms = 0;
#pragma unroll
      for (int i = 1; i < 8; ++i)
        if (cd[i] < mv) { mv = cd[i]; mt = cw[i]; ms = i; }
      int mi = lane * 8 + ms;
#pragma unroll
      for (int off = 16; off > 0; off >>= 1) {
        const float ov = __shfl_xor(mv, off);
        const float ot = __shfl_xor(mt, off);
        const int   oi = __shfl_xor(mi, off);
        if (ov < mv || (ov == mv && oi < mi)) { mv = ov; mt = ot; mi = oi; }
      }
#pragma unroll
      for (int i = 0; i < 8; ++i)
        if (mi == lane * 8 + i) cd[i] = 3.0e38f;  // invalidate winner
      const float w = 1.0f / (mv + 1e-4f);
      num += w * mt;
      den += w;
    }
    den = fmaxf(den, 1e-9f);
    if (lane == 0) out[qbase + q] = num / den;
  }
}

// ---------------------------------------------------------------------------
extern "C" void kernel_launch(void* const* d_in, const int* in_sizes, int n_in,
                              void* d_out, int out_size, void* d_ws, size_t ws_size,
                              hipStream_t stream) {
  const float* points   = (const float*)d_in[0];  // [8,2048,64]
  const float* features = (const float*)d_in[1];  // [20000,64]
  const float* targets  = (const float*)d_in[2];  // [20000]
  float* out = (float*)d_out;                     // [16384]

  char* ws = (char*)d_ws;
  _Float16* qf16 = (_Float16*)(ws + OFF_QF16);
  _Float16* ff16 = (_Float16*)(ws + OFF_FF16);
  float*    qn   = (float*)(ws + OFF_QN);
  float*    fn   = (float*)(ws + OFF_FN);

  prep_kernel<<<(NQ + NF) / 8, 256, 0, stream>>>(points, features, qf16, ff16, qn, fn);
  knn_wmma_kernel<<<NQ / 16, 256, 0, stream>>>(qf16, ff16, qn, fn, targets, out);
}
